// LoRAAttnProcessor_9929964389038
// MI455X (gfx1250) — matrix operand order repaired and checked by reading, not compiled, not run on hardware
//
#include <hip/hip_runtime.h>
#include <hip/hip_bf16.h>

// ---------------------------------------------------------------------------
// LoRA attention (B=4, S=2048, H=1024, heads=16, d=64, rank=8) for gfx1250.
// bf16 WMMA (v_wmma_f32_16x16x32_bf16) for all GEMMs + flash attention.
// GEMM: ping-pong double buffer (unroll-by-2, no register rotation copies).
// ---------------------------------------------------------------------------

typedef __bf16 bf16;
typedef bf16  v16bf __attribute__((ext_vector_type(16)));
typedef bf16  v8bf  __attribute__((ext_vector_type(8)));
typedef float v8f   __attribute__((ext_vector_type(8)));

#define WMMA_BF16(a, b, c)                                                    \
  __builtin_amdgcn_wmma_f32_16x16x32_bf16(false, (a), false, (b), (short)0,   \
                                          (c), false, false)

static constexpr int Bn     = 4;
static constexpr int S      = 2048;
static constexpr int H      = 1024;
static constexpr int HEADS  = 16;
static constexpr int DHEAD  = 64;
static constexpr int RANK   = 8;
static constexpr int ROWS   = Bn * S;      // 8192
static constexpr int N_QKV  = 3 * H;       // 3072

// --- Fragment loaders for 16-bit WMMA, contiguous k order shared by A and B ----------
// A matrix 16x32 (M x K), per lane: M = lane&15,
//   element j: K = (lane>=16 ? 16 : 0) + j
__device__ __forceinline__ v16bf load_a_frag(const bf16* __restrict__ base,
                                             int ld, int row0, int k0) {
  const int lane = threadIdx.x & 31;
  const int l16  = lane & 15;
  const int hi   = lane >> 4;
  const bf16* p  = base + (size_t)(row0 + l16) * ld + k0 + hi * 16;
  v8bf lo = *(const v8bf*)(p);
  v8bf up = *(const v8bf*)(p + 8);
  v16bf a;
#pragma unroll
  for (int j = 0; j < 8; ++j) { a[j] = lo[j]; a[j + 8] = up[j]; }
  return a;
}

// B matrix 32x16 (K x N), per lane: N = lane&15,
//   element j: K = (lane>=16 ? 16 : 0) + j
// Source is row-major [N_rows x ld]: row picks N, K runs contiguously.
__device__ __forceinline__ v16bf load_b_frag(const bf16* __restrict__ base,
                                             int ld, int n0, int k0) {
  const int lane = threadIdx.x & 31;
  const int l16  = lane & 15;
  const int hi   = lane >> 4;
  const bf16* p  = base + (size_t)(n0 + l16) * ld + k0 + hi * 16;
  v8bf lo = *(const v8bf*)(p);
  v8bf up = *(const v8bf*)(p + 8);
  v16bf b;
#pragma unroll
  for (int j = 0; j < 8; ++j) { b[j] = lo[j]; b[j + 8] = up[j]; }
  return b;
}

// --- fp32 -> bf16 conversion ------------------------------------------------
__global__ __launch_bounds__(256) void cvt_f32_bf16(
    const float* __restrict__ in, bf16* __restrict__ out, int n) {
  int i = blockIdx.x * 256 + threadIdx.x;
  if (i < n) out[i] = (bf16)in[i];
}

// --- GEMM: C[M x N] = A[M x K](bf16) @ W[N x K](bf16)^T (+bias) ------------
// 256 threads = 8 waves arranged 4(M) x 2(N); wave tile 32M x 64N.
// Block tile 128M x 128N. Ping-pong double-buffered k-loop (step 64):
// compute on one fragment set while the other set's loads are in flight.
struct Frags {
  v16bf a[2];
  v16bf b[4];
};

__device__ __forceinline__ void load_frags(Frags& f,
                                           const bf16* __restrict__ A,
                                           const bf16* __restrict__ W,
                                           int K, int m0, int n0, int k0) {
#pragma unroll
  for (int im = 0; im < 2; ++im) f.a[im] = load_a_frag(A, K, m0 + im * 16, k0);
#pragma unroll
  for (int t = 0; t < 4; ++t)    f.b[t]  = load_b_frag(W, K, n0 + t * 16, k0);
}

__device__ __forceinline__ void mma8(v8f c[2][4], const Frags& f) {
#pragma unroll
  for (int im = 0; im < 2; ++im)
#pragma unroll
    for (int t = 0; t < 4; ++t)
      c[im][t] = WMMA_BF16(f.a[im], f.b[t], c[im][t]);
}

__global__ __launch_bounds__(256) void gemm_xwT(
    const bf16* __restrict__ A, const bf16* __restrict__ W,
    float* __restrict__ C, const float* __restrict__ bias,
    int M, int N, int K) {
  const int wave = threadIdx.x >> 5;
  const int wm   = wave >> 1;           // 0..3
  const int wn   = wave & 1;            // 0..1
  const int m0   = blockIdx.y * 128 + wm * 32;
  const int n0   = blockIdx.x * 128 + wn * 64;
  const int lane = threadIdx.x & 31;
  const int l16  = lane & 15;
  const int hi   = lane >> 4;

  v8f c[2][4];
#pragma unroll
  for (int im = 0; im < 2; ++im)
#pragma unroll
    for (int t = 0; t < 4; ++t) c[im][t] = v8f{};

  Frags f0, f1;
  load_frags(f0, A, W, K, m0, n0, 0);

  // K is a multiple of 64: 2 k-steps per iteration, ping-pong f0/f1.
  int k0 = 0;
  for (; k0 < K - 64; k0 += 64) {
    // Prefetch two double-steps ahead into L2 (global_prefetch_b8).
    if (k0 + 128 < K) {
      __builtin_prefetch(A + (size_t)(m0 + l16) * K + k0 + 128, 0, 3);
      __builtin_prefetch(A + (size_t)(m0 + 16 + l16) * K + k0 + 128, 0, 3);
      __builtin_prefetch(W + (size_t)(n0 + l16) * K + k0 + 128, 0, 3);
      __builtin_prefetch(W + (size_t)(n0 + 16 + l16) * K + k0 + 128, 0, 3);
    }
    load_frags(f1, A, W, K, m0, n0, k0 + 32);  // fill f1 ...
    mma8(c, f0);                               // ... while computing on f0
    load_frags(f0, A, W, K, m0, n0, k0 + 64);  // fill f0 ...
    mma8(c, f1);                               // ... while computing on f1
  }
  // Tail pair: k0 == K - 64.
  load_frags(f1, A, W, K, m0, n0, k0 + 32);
  mma8(c, f0);
  mma8(c, f1);

#pragma unroll
  for (int im = 0; im < 2; ++im)
#pragma unroll
    for (int t = 0; t < 4; ++t) {
      int col  = n0 + t * 16 + l16;
      float bv = bias ? bias[col] : 0.0f;
#pragma unroll
      for (int v = 0; v < 8; ++v) {
        int row = m0 + im * 16 + v + hi * 8;
        C[(size_t)row * N + col] = c[im][t][v] + bv;
      }
    }
}

// --- LoRA (rank 8 on the projection OUTPUT) + bf16 convert + layout --------
// One block per sequence row s. y[h] += sum_r (sum_h' y[h']*D[r][h']) * U[h][r]
__global__ __launch_bounds__(256) void lora_convert(
    const float* __restrict__ Y, int ldY, int colOff,
    const float* __restrict__ D,   // [RANK][H]
    const float* __restrict__ U,   // [H][RANK]
    bf16* __restrict__ out, int vtranspose) {
  __shared__ float red[RANK * 256];
  __shared__ float tf[RANK];
  const int s   = blockIdx.x;
  const int tid = threadIdx.x;
  const int h0  = tid * 4;
  const float* yrow = Y + (size_t)s * ldY + colOff;

  float y[4];
#pragma unroll
  for (int i = 0; i < 4; ++i) y[i] = yrow[h0 + i];

#pragma unroll
  for (int r = 0; r < RANK; ++r) {
    const float* dr = D + r * H;
    float acc = 0.0f;
#pragma unroll
    for (int i = 0; i < 4; ++i) acc += y[i] * dr[h0 + i];
    red[r * 256 + tid] = acc;
  }
  __syncthreads();
  for (int st = 128; st > 0; st >>= 1) {
    if (tid < st) {
#pragma unroll
      for (int r = 0; r < RANK; ++r)
        red[r * 256 + tid] += red[r * 256 + tid + st];
    }
    __syncthreads();
  }
  if (tid < RANK) tf[tid] = red[tid * 256];
  __syncthreads();

#pragma unroll
  for (int i = 0; i < 4; ++i) {
    int hcol = h0 + i;
    float val = y[i];
#pragma unroll
    for (int r = 0; r < RANK; ++r) val += tf[r] * U[hcol * RANK + r];
    if (!vtranspose) {
      out[(size_t)s * H + hcol] = (bf16)val;
    } else {
      int b = s >> 11, srow = s & (S - 1);         // S = 2048
      out[((size_t)b * H + hcol) * S + srow] = (bf16)val;  // Vt[b][h][s]
    }
  }
}

// --- Flash attention: softmax(Q K^T / 8) V per (batch, head) ---------------
// 128 threads = 4 waves; each wave owns 32 q-rows (two 16-row tiles sharing
// K/V fragments); k-loop in chunks of 32 keys: 16 WMMA per 8 fragment loads.
__global__ __launch_bounds__(128) void flash_attn(
    const bf16* __restrict__ Qb,   // [ROWS][H] row-major
    const bf16* __restrict__ Kb,   // [ROWS][H] row-major
    const bf16* __restrict__ Vt,   // [Bn][H][S]  (transposed V)
    bf16* __restrict__ Ob) {       // [ROWS][H] row-major
  const int bh   = blockIdx.x;     // b*HEADS + h
  const int qblk = blockIdx.y;     // 128-row q block
  const int b    = bh >> 4;
  const int h    = bh & 15;
  const int wave = threadIdx.x >> 5;
  const int lane = threadIdx.x & 31;
  const int l16  = lane & 15;
  const int hi   = lane >> 4;
  const int qrow0 = qblk * 128 + wave * 32;

  const bf16* Qbase = Qb + (size_t)(b * S) * H + h * DHEAD;
  const bf16* Kbase = Kb + (size_t)(b * S) * H + h * DHEAD;
  const bf16* Vbase = Vt + (size_t)b * H * S + (size_t)h * DHEAD * S;

  __shared__ float pbuf[4][32][34];  // per-wave P staging (C-layout -> A-layout)

  // Q fragments are loop-invariant: 2 q-tiles x 2 d-chunks.
  v16bf qa[2][2];
#pragma unroll
  for (int qt = 0; qt < 2; ++qt) {
    qa[qt][0] = load_a_frag(Qbase, H, qrow0 + qt * 16, 0);
    qa[qt][1] = load_a_frag(Qbase, H, qrow0 + qt * 16, 32);
  }

  v8f acc[2][4];
  float mrow[2][8], lrow[2][8];
#pragma unroll
  for (int qt = 0; qt < 2; ++qt) {
#pragma unroll
    for (int t = 0; t < 4; ++t) acc[qt][t] = v8f{};
#pragma unroll
    for (int v = 0; v < 8; ++v) { mrow[qt][v] = -1e30f; lrow[qt][v] = 0.0f; }
  }
  const float scale = 0.125f;        // 1/sqrt(64)

  for (int kk = 0; kk < S; kk += 32) {
    // ---- K fragments for this 32-key chunk (shared by both q-tiles) ----
    v16bf kf[2][2];                  // [col-chunk][d-chunk]
    kf[0][0] = load_b_frag(Kbase, H, kk,      0);
    kf[0][1] = load_b_frag(Kbase, H, kk,      32);
    kf[1][0] = load_b_frag(Kbase, H, kk + 16, 0);
    kf[1][1] = load_b_frag(Kbase, H, kk + 16, 32);
    if (kk + 32 < S) {               // prefetch next chunk's K rows
      __builtin_prefetch(Kbase + (size_t)(kk + 32 + l16) * H, 0, 3);
      __builtin_prefetch(Kbase + (size_t)(kk + 48 + l16) * H, 0, 3);
    }
    // ---- scores: 2 q-tiles x 2 col-chunks, K-dim = d = 64 ----
    v8f sc[2][2];
#pragma unroll
    for (int qt = 0; qt < 2; ++qt)
#pragma unroll
      for (int cc = 0; cc < 2; ++cc) {
        v8f s = v8f{};
        s = WMMA_BF16(qa[qt][0], kf[cc][0], s);
        s = WMMA_BF16(qa[qt][1], kf[cc][1], s);
        sc[qt][cc] = s;
      }
    // ---- V fragments now: their latency hides under the softmax VALU ----
    v16bf vb[4];
#pragma unroll
    for (int t = 0; t < 4; ++t) vb[t] = load_b_frag(Vbase, S, t * 16, kk);
    // ---- online softmax (rows 0-7 in lanes 0-15, rows 8-15 in 16-31) ----
#pragma unroll
    for (int qt = 0; qt < 2; ++qt) {
#pragma unroll
      for (int v = 0; v < 8; ++v) {
        float sa = sc[qt][0][v] * scale, sb = sc[qt][1][v] * scale;
        float mv = fmaxf(sa, sb);
#pragma unroll
        for (int off = 1; off < 16; off <<= 1)
          mv = fmaxf(mv, __shfl_xor(mv, off, 16));
        float mnew  = fmaxf(mrow[qt][v], mv);
        float alpha = __expf(mrow[qt][v] - mnew);
        mrow[qt][v] = mnew;
        float p0 = __expf(sa - mnew);
        float p1 = __expf(sb - mnew);
        sc[qt][0][v] = p0; sc[qt][1][v] = p1;
        float ls = p0 + p1;
#pragma unroll
        for (int off = 1; off < 16; off <<= 1)
          ls += __shfl_xor(ls, off, 16);
        lrow[qt][v] = lrow[qt][v] * alpha + ls;
#pragma unroll
        for (int t = 0; t < 4; ++t) acc[qt][t][v] *= alpha;
      }
    }
    // ---- stage P (C-layout) to LDS, both q-tiles ----
#pragma unroll
    for (int qt = 0; qt < 2; ++qt)
#pragma unroll
      for (int v = 0; v < 8; ++v) {
        int r = qt * 16 + v + hi * 8;
        pbuf[wave][r][l16]      = sc[qt][0][v];
        pbuf[wave][r][16 + l16] = sc[qt][1][v];
      }
    asm volatile("s_wait_dscnt 0" ::: "memory");
    // ---- rebuild P as A fragments (16x32 bf16) and do PV ----
#pragma unroll
    for (int qt = 0; qt < 2; ++qt) {
      v16bf pa;
#pragma unroll
      for (int j = 0; j < 16; ++j) {
        int kd = hi * 16 + j;
        pa[j] = (bf16)pbuf[wave][qt * 16 + l16][kd];
      }
#pragma unroll
      for (int t = 0; t < 4; ++t)
        acc[qt][t] = WMMA_BF16(pa, vb[t], acc[qt][t]);
    }
  }
  // ---- epilogue: normalize and store bf16 [s][h*64+d] ----
#pragma unroll
  for (int qt = 0; qt < 2; ++qt)
#pragma unroll
    for (int v = 0; v < 8; ++v) {
      float inv = 1.0f / lrow[qt][v];
      int row = qrow0 + qt * 16 + v + hi * 8;
      bf16* orow = Ob + (size_t)(b * S + row) * H + h * DHEAD;
#pragma unroll
      for (int t = 0; t < 4; ++t)
        orow[t * 16 + l16] = (bf16)(acc[qt][t][v] * inv);
    }
}

// ---------------------------------------------------------------------------
extern "C" void kernel_launch(void* const* d_in, const int* in_sizes, int n_in,
                              void* d_out, int out_size, void* d_ws,
                              size_t ws_size, hipStream_t stream) {
  (void)in_sizes; (void)n_in; (void)out_size; (void)ws_size;

  const float* x  = (const float*)d_in[0];
  const float* Wq = (const float*)d_in[1];
  const float* Wk = (const float*)d_in[2];
  const float* Wv = (const float*)d_in[3];
  const float* Dq = (const float*)d_in[4];
  const float* Uq = (const float*)d_in[5];
  const float* Dk = (const float*)d_in[6];
  const float* Uk = (const float*)d_in[7];
  const float* Dv = (const float*)d_in[8];
  const float* Uv = (const float*)d_in[9];
  const float* Wo = (const float*)d_in[10];
  const float* bo = (const float*)d_in[11];
  float* out = (float*)d_out;

  // Workspace carve-up (16B-aligned by construction)
  char* ws = (char*)d_ws;
  size_t off = 0;
  bf16* xb    = (bf16*)(ws + off);  off += (size_t)ROWS * H * 2;        // 16 MB
  bf16* wqkvb = (bf16*)(ws + off);  off += (size_t)N_QKV * H * 2;       //  6 MB
  bf16* wob   = (bf16*)(ws + off);  off += (size_t)H * H * 2;           //  2 MB
  float* yqkv = (float*)(ws + off); off += (size_t)ROWS * N_QKV * 4;    // 96 MB
  bf16* qb    = (bf16*)(ws + off);  off += (size_t)ROWS * H * 2;        // 16 MB
  bf16* kb    = (bf16*)(ws + off);  off += (size_t)ROWS * H * 2;        // 16 MB
  bf16* vt    = (bf16*)(ws + off);  off += (size_t)Bn * H * S * 2;      // 16 MB
  bf16* ob    = (bf16*)(ws + off);  off += (size_t)ROWS * H * 2;        // 16 MB

  // 1) fp32 -> bf16 conversions
  {
    int n = ROWS * H;
    cvt_f32_bf16<<<(n + 255) / 256, 256, 0, stream>>>(x, xb, n);
    int nw = H * H;
    cvt_f32_bf16<<<(nw + 255) / 256, 256, 0, stream>>>(Wq, wqkvb, nw);
    cvt_f32_bf16<<<(nw + 255) / 256, 256, 0, stream>>>(Wk, wqkvb + (size_t)H * H, nw);
    cvt_f32_bf16<<<(nw + 255) / 256, 256, 0, stream>>>(Wv, wqkvb + (size_t)2 * H * H, nw);
    cvt_f32_bf16<<<(nw + 255) / 256, 256, 0, stream>>>(Wo, wob, nw);
  }

  // 2) fused QKV projection: yqkv[8192 x 3072] = xb @ wqkvb^T
  gemm_xwT<<<dim3(N_QKV / 128, ROWS / 128), 256, 0, stream>>>(
      xb, wqkvb, yqkv, nullptr, ROWS, N_QKV, H);

  // 3) LoRA on projection outputs + bf16 convert + layout for attention
  lora_convert<<<ROWS, 256, 0, stream>>>(yqkv, N_QKV, 0 * H, Dq, Uq, qb, 0);
  lora_convert<<<ROWS, 256, 0, stream>>>(yqkv, N_QKV, 1 * H, Dk, Uk, kb, 0);
  lora_convert<<<ROWS, 256, 0, stream>>>(yqkv, N_QKV, 2 * H, Dv, Uv, vt, 1);

  // 4) flash attention per (batch, head), 128 q-rows per block
  flash_attn<<<dim3(Bn * HEADS, S / 128), 128, 0, stream>>>(qb, kb, vt, ob);

  // 5) output projection with bias -> d_out (fp32)
  gemm_xwT<<<dim3(H / 128, ROWS / 128), 256, 0, stream>>>(
      ob, wob, out, bo, ROWS, H, H);
}
